// TdnnLiGRU_19842748907934
// MI455X (gfx1250) — matrix operand order repaired and checked by reading, not compiled, not run on hardware
//
#include <hip/hip_runtime.h>
#include <hip/hip_bf16.h>

// ---------------------------------------------------------------------------
// TDNN-LiGRU for MI455X (gfx1250), wave32 + WMMA bf16 + async global->LDS.
//  B=32, T=512, D=H=1024, E=2H=2048, L=3
//  Phase 1 per layer: w = BN(x @ W^T)  -- 16384x2048x1024 GEMM, bf16 WMMA,
//                     double-buffered LDS fed by GLOBAL_LOAD_ASYNC_TO_LDS.
//  Phase 2 per layer: persistent LiGRU scan, U slice resident in LDS,
//                     grid-wide barrier per time step.
// ---------------------------------------------------------------------------

#define B_ 32
#define T_ 512
#define D_ 1024
#define H_ 1024
#define E_ 2048
#define L_ 3
#define M_ (B_ * T_)   // 16384

typedef __bf16 v16bf __attribute__((ext_vector_type(16)));
typedef float  v8f   __attribute__((ext_vector_type(8)));

struct Frag32 { uint4 lo, hi; };   // 32 bytes -> v16bf

__device__ __forceinline__ unsigned short f32_to_bf16(float f) {
  unsigned u = __float_as_uint(f);
  unsigned r = 0x7FFFu + ((u >> 16) & 1u);      // round-to-nearest-even
  return (unsigned short)((u + r) >> 16);
}
__device__ __forceinline__ float bf16_to_f32(unsigned short h) {
  return __uint_as_float(((unsigned)h) << 16);
}

// --- CDNA5 async global->LDS copy (16B per lane), tracked by ASYNCcnt -------
__device__ __forceinline__ void async_g2l_b128(unsigned lds_off, const void* gaddr) {
  asm volatile("global_load_async_to_lds_b128 %0, %1, off"
               :: "v"(lds_off), "v"(gaddr) : "memory");
}
__device__ __forceinline__ void wait_async0() {
  asm volatile("s_wait_asynccnt 0x0" ::: "memory");
}
// LDS byte offset of a shared-memory pointer (generic addr low 32 bits).
__device__ __forceinline__ unsigned lds_off_of(const void* p) {
  return (unsigned)(unsigned long long)p;
}

// A-matrix fragment (16x32 bf16), ISA 7.12.2 layout:
// lane l: m=l&15, hi=l>>4; elem i: k = 8*hi + (i%8) + 16*(i/8)
__device__ __forceinline__ v16bf frag_a(const unsigned short* base, int rowStride,
                                        int koff, int lane) {
  const int r = lane & 15, hi = lane >> 4;
  const unsigned short* row = base + (size_t)r * rowStride + koff;
  Frag32 f;
  f.lo = *(const uint4*)(row + hi * 8);
  f.hi = *(const uint4*)(row + 16 + hi * 8);
  return __builtin_bit_cast(v16bf, f);
}

// B-matrix fragment (32x16 bf16), staged as [n][k] rows:
// lane l: n=l&15, hi=l>>4; elem i: k = 8*hi + (i%8) + 16*(i/8)
__device__ __forceinline__ v16bf frag_b(const unsigned short* base, int rowStride,
                                        int koff, int lane) {
  const int r = lane & 15, hi = lane >> 4;
  const unsigned short* row = base + (size_t)r * rowStride + koff;
  Frag32 f;
  f.lo = *(const uint4*)(row + hi * 8);
  f.hi = *(const uint4*)(row + 16 + hi * 8);
  return __builtin_bit_cast(v16bf, f);
}

// ---------------------------------------------------------------------------
// utility kernels
// ---------------------------------------------------------------------------
__global__ void k_f32_to_bf16(const float* __restrict__ in,
                              unsigned short* __restrict__ out, int n) {
  int i = blockIdx.x * blockDim.x + threadIdx.x;
  int stride = gridDim.x * blockDim.x;
  for (; i < n; i += stride) out[i] = f32_to_bf16(in[i]);
}

__global__ void k_init(unsigned* counters) {
  if (threadIdx.x < 8) counters[threadIdx.x] = 0u;
}

// ---------------------------------------------------------------------------
// Phase 1: w_bf16[m,e] = BN(x @ W^T)   (M_ x E_ x K=1024)
// 256 threads = 8 waves; block tile 128x128, BK=32; wave tile 64x32 (4x2 frags)
// Double-buffered LDS; next K-tile staged via async copies during WMMAs.
// ---------------------------------------------------------------------------
__global__ __launch_bounds__(256)
void k_gemm_bn(const unsigned short* __restrict__ X,    // (M_, 1024) bf16
               const unsigned short* __restrict__ Wl,   // (E_, 1024) bf16
               const float* __restrict__ gamma, const float* __restrict__ beta,
               const float* __restrict__ mean,  const float* __restrict__ var,
               unsigned short* __restrict__ Wout) {     // (M_, E_) bf16
  __shared__ unsigned short As[2][128 * 32];   // 2 x 8 KB
  __shared__ unsigned short Bs[2][128 * 32];   // 2 x 8 KB

  const int m0 = blockIdx.x * 128;
  const int n0 = blockIdx.y * 128;
  const int tid = threadIdx.x;
  const int lane = tid & 31, wave = tid >> 5;
  const int waveM = wave >> 2, waveN = wave & 3;

  const unsigned asBase = lds_off_of(&As[0][0]);
  const unsigned bsBase = lds_off_of(&Bs[0][0]);

  // stage one 128x32 K-tile of A and B into LDS buffer `buf` (async)
  auto stage = [&](int buf, int k0) {
#pragma unroll
    for (int it = 0; it < 2; ++it) {
      int c = tid + it * 256;                  // 512 16B-chunks per tile
      int row = c >> 2, q = c & 3;
      unsigned doff = (unsigned)(buf * 8192 + (row * 32 + q * 8) * 2);
      async_g2l_b128(asBase + doff, X  + (size_t)(m0 + row) * D_ + k0 + q * 8);
      async_g2l_b128(bsBase + doff, Wl + (size_t)(n0 + row) * D_ + k0 + q * 8);
    }
  };

  v8f acc[4][2];
#pragma unroll
  for (int i = 0; i < 4; ++i)
#pragma unroll
    for (int j = 0; j < 2; ++j) acc[i][j] = (v8f){0.f,0.f,0.f,0.f,0.f,0.f,0.f,0.f};

  stage(0, 0);
  for (int kt = 0; kt < 32; ++kt) {
    wait_async0();          // this wave's copies into buf[kt&1] are done
    __syncthreads();        // everyone's copies are done; prior reads retired
    if (kt + 1 < 32) stage((kt + 1) & 1, (kt + 1) * 32);

    const unsigned short* Ab = As[kt & 1];
    const unsigned short* Bb = Bs[kt & 1];
    v16bf af[4], bfr[2];
#pragma unroll
    for (int i = 0; i < 4; ++i)
      af[i] = frag_a(Ab + (waveM * 64 + i * 16) * 32, 32, 0, lane);
#pragma unroll
    for (int j = 0; j < 2; ++j)
      bfr[j] = frag_b(Bb + (waveN * 32 + j * 16) * 32, 32, 0, lane);

#pragma unroll
    for (int i = 0; i < 4; ++i)
#pragma unroll
      for (int j = 0; j < 2; ++j)
        acc[i][j] = __builtin_amdgcn_wmma_f32_16x16x32_bf16(
            false, af[i], false, bfr[j], (short)0, acc[i][j], false, false);
  }

  // epilogue: BN affine then bf16 store
  const int hi = lane >> 4, nl = lane & 15;
#pragma unroll
  for (int j = 0; j < 2; ++j) {
    int e = n0 + waveN * 32 + j * 16 + nl;
    float sc = gamma[e] * rsqrtf(var[e] + 1e-5f);
    float mu = mean[e], bt = beta[e];
#pragma unroll
    for (int i = 0; i < 4; ++i) {
      int mbase = m0 + waveM * 64 + i * 16 + hi * 8;
#pragma unroll
      for (int r = 0; r < 8; ++r) {
        float v = (acc[i][j][r] - mu) * sc + bt;
        Wout[(size_t)(mbase + r) * E_ + e] = f32_to_bf16(v);
      }
    }
  }
}

// ---------------------------------------------------------------------------
// Phase 2: persistent LiGRU scan.
// 64 WGs x 128 threads (4 waves). WG g owns h columns [g*16, g*16+16).
// LDS: U slice 32x1024 bf16 (64 KB, resident all 512 steps)
//      h stage 32x1024 bf16 (64 KB, refreshed per step via async copies)
//      Gs: 4 x 16x16 f32 accumulator tiles (4 KB)
// Grid barrier: device-scope atomic counter + s_sleep spin.
// ---------------------------------------------------------------------------
#define RNN_NWG 64
#define RNN_THREADS 128
#define RNN_SMEM_BYTES (2 * 32 * 1024 * 2 + 4 * 256 * 4)

__device__ __forceinline__ void grid_sync(unsigned* cnt, unsigned target) {
  __syncthreads();
  if (threadIdx.x == 0) {
    __hip_atomic_fetch_add(cnt, 1u, __ATOMIC_ACQ_REL, __HIP_MEMORY_SCOPE_AGENT);
    while (__hip_atomic_load(cnt, __ATOMIC_ACQUIRE, __HIP_MEMORY_SCOPE_AGENT) < target) {
      __builtin_amdgcn_s_sleep(1);
    }
  }
  __syncthreads();
}

__global__ __launch_bounds__(RNN_THREADS)
void k_ligru_scan(const unsigned short* __restrict__ Wg,  // (M_, E_) pre-gates bf16
                  const unsigned short* __restrict__ Ul,  // (E_, H_) bf16
                  unsigned short* __restrict__ Xnext,     // (M_, H_) bf16 hs out
                  float* __restrict__ Xout,               // (M_, H_) f32 hs out or null
                  float* __restrict__ HH,                 // d_out hh, pre-offset by l*H_
                  float* __restrict__ Hf,                 // (B_, H_) f32 state
                  unsigned short* __restrict__ Hb,        // (2, B_, H_) bf16 state
                  unsigned* __restrict__ barrier_cnt) {
  extern __shared__ unsigned short smem[];
  unsigned short* Us = smem;                   // 32*1024 shorts
  unsigned short* Hs = smem + 32 * 1024;       // 32*1024 shorts
  float* Gs = (float*)(smem + 64 * 1024);      // 4*256 floats

  const int g = blockIdx.x;
  const int j0 = g * 16;
  const int tid = threadIdx.x, lane = tid & 31, wave = tid >> 5;
  const int mt = wave & 1, gate = wave >> 1;
  const unsigned usBase = lds_off_of(Us);
  const unsigned hsBase = lds_off_of(Hs);

  // stage U slice once (async): rows 0..15 = U[j0+r,:], 16..31 = U[H_+j0+r-16,:]
#pragma unroll
  for (int i = 0; i < 32; ++i) {
    int c = tid + i * RNN_THREADS;             // 4096 chunks of 16B
    int r = c >> 7, q = c & 127;
    int eg = (r < 16) ? (j0 + r) : (H_ + j0 + r - 16);
    async_g2l_b128(usBase + (unsigned)c * 16, Ul + (size_t)eg * H_ + q * 8);
  }

  // h := 0 for owned columns (both ping-pong buffers)
#pragma unroll
  for (int k = 0; k < 4; ++k) {
    int i = tid + k * RNN_THREADS;             // 0..511
    int b = i >> 4, n = i & 15;
    Hf[b * H_ + j0 + n] = 0.f;
    Hb[b * H_ + j0 + n] = 0;
    Hb[B_ * H_ + b * H_ + j0 + n] = 0;
  }
  __threadfence();
  wait_async0();                               // U resident before first use
  unsigned bar = 0;
  grid_sync(barrier_cnt, ++bar * RNN_NWG);

  for (int t = 0; t < T_; ++t) {
    // stage full h (32x1024 bf16) from the "read" ping-pong buffer (async)
    const unsigned short* hsrc = Hb + (size_t)(t & 1) * (B_ * H_);
#pragma unroll
    for (int i = 0; i < 32; ++i) {
      int c = tid + i * RNN_THREADS;
      int b = c >> 7, q = c & 127;
      async_g2l_b128(hsBase + (unsigned)c * 16, hsrc + (size_t)b * H_ + q * 8);
    }
    wait_async0();
    __syncthreads();

    // acc = h[mt rows] @ U[gate rows]^T over K=1024 (32 chained WMMAs)
    v8f acc = (v8f){0.f,0.f,0.f,0.f,0.f,0.f,0.f,0.f};
    const unsigned short* Arow = Hs + mt * 16 * 1024;
    const unsigned short* Brow = Us + gate * 16 * 1024;
#pragma unroll 4
    for (int kk = 0; kk < 32; ++kk) {
      v16bf a = frag_a(Arow, 1024, kk * 32, lane);
      v16bf b = frag_b(Brow, 1024, kk * 32, lane);
      acc = __builtin_amdgcn_wmma_f32_16x16x32_bf16(false, a, false, b,
                                                    (short)0, acc, false, false);
    }
    // dump accumulator tile to LDS for gate pairing
    {
      int m = (lane >> 4) * 8, n = lane & 15;
#pragma unroll
      for (int r = 0; r < 8; ++r) Gs[wave * 256 + (m + r) * 16 + n] = acc[r];
    }
    __syncthreads();

    // elementwise LiGRU update on owned 32x16 slice
#pragma unroll
    for (int k = 0; k < 4; ++k) {
      int i = tid + k * RNN_THREADS;
      int b = i >> 4, n = i & 15;
      int mtl = b >> 4, ml = b & 15;
      size_t grow = ((size_t)b * T_ + t) * E_;
      float av = Gs[mtl * 256 + ml * 16 + n]       + bf16_to_f32(Wg[grow + j0 + n]);
      float zv = Gs[(2 + mtl) * 256 + ml * 16 + n] + bf16_to_f32(Wg[grow + H_ + j0 + n]);
      float z = 1.0f / (1.0f + __expf(-zv));
      float hc = fmaxf(av, 0.0f);
      float hold = Hf[b * H_ + j0 + n];
      float hnew = z * hold + (1.0f - z) * hc;
      Hf[b * H_ + j0 + n] = hnew;
      unsigned short hb = f32_to_bf16(hnew);
      Hb[(size_t)((t + 1) & 1) * (B_ * H_) + b * H_ + j0 + n] = hb;
      Xnext[((size_t)b * T_ + t) * H_ + j0 + n] = hb;
      if (Xout) Xout[((size_t)b * T_ + t) * H_ + j0 + n] = hnew;
      if (t == T_ - 1) HH[(size_t)b * (L_ * H_) + j0 + n] = hnew;
    }
    __threadfence();
    grid_sync(barrier_cnt, ++bar * RNN_NWG);
  }
}

// ---------------------------------------------------------------------------
// host-side orchestration
// ---------------------------------------------------------------------------
extern "C" void kernel_launch(void* const* d_in, const int* in_sizes, int n_in,
                              void* d_out, int out_size, void* d_ws, size_t ws_size,
                              hipStream_t stream) {
  (void)in_sizes; (void)n_in; (void)out_size; (void)ws_size;
  const float* x     = (const float*)d_in[0];
  const float* W     = (const float*)d_in[1];
  const float* U     = (const float*)d_in[2];
  const float* gamma = (const float*)d_in[3];
  const float* beta  = (const float*)d_in[4];
  const float* mean  = (const float*)d_in[5];
  const float* var   = (const float*)d_in[6];

  float* out_x  = (float*)d_out;                       // (B,T,H) f32
  float* out_hh = out_x + (size_t)B_ * T_ * H_;        // (B,L,H) f32

  // workspace carve-up (~126 MB total)
  char* ws = (char*)d_ws;
  unsigned* counters = (unsigned*)ws;                                  size_t off = 256;
  unsigned short* xb = (unsigned short*)(ws + off);  off += (size_t)M_ * D_ * 2;       // 32 MB
  unsigned short* wb = (unsigned short*)(ws + off);  off += (size_t)M_ * E_ * 2;       // 64 MB
  unsigned short* Wb = (unsigned short*)(ws + off);  off += (size_t)L_ * E_ * D_ * 2;  // 12 MB
  unsigned short* Ub = (unsigned short*)(ws + off);  off += (size_t)L_ * E_ * D_ * 2;  // 12 MB
  float*          Hf = (float*)(ws + off);           off += (size_t)B_ * H_ * 4;
  unsigned short* Hb = (unsigned short*)(ws + off);  off += (size_t)2 * B_ * H_ * 2;

  k_init<<<1, 32, 0, stream>>>(counters);
  k_f32_to_bf16<<<1024, 256, 0, stream>>>(x, xb, M_ * D_);
  k_f32_to_bf16<<<1024, 256, 0, stream>>>(W, Wb, L_ * E_ * D_);
  k_f32_to_bf16<<<1024, 256, 0, stream>>>(U, Ub, L_ * E_ * D_);

  for (int l = 0; l < L_; ++l) {
    k_gemm_bn<<<dim3(M_ / 128, E_ / 128), 256, 0, stream>>>(
        xb, Wb + (size_t)l * E_ * D_,
        gamma + l * E_, beta + l * E_, mean + l * E_, var + l * E_, wb);
    k_ligru_scan<<<RNN_NWG, RNN_THREADS, RNN_SMEM_BYTES, stream>>>(
        wb, Ub + (size_t)l * E_ * D_,
        xb,                                   // hs (bf16) becomes next layer's x
        (l == L_ - 1) ? out_x : nullptr,      // f32 hs only for last layer
        out_hh + (size_t)l * H_,
        Hf, Hb, counters + l);
  }
}
